// Encoder_45268955300430
// MI455X (gfx1250) — compile-verified, matrix operand order checked by reading, not run on hardware
//
#include <hip/hip_runtime.h>
#include <hip/hip_bf16.h>

typedef __bf16 bf16_t;
typedef __attribute__((ext_vector_type(16))) __bf16 v16bf;
typedef __attribute__((ext_vector_type(8)))  __bf16 v8bf;
typedef __attribute__((ext_vector_type(8)))  float  v8f;
typedef __attribute__((ext_vector_type(4)))  float  v4f;

#define DIM   128
#define NGRP  256
#define GINW  1.1f      // 1 + eps_gin
#define BNEPS 1e-5f
#define WPAD  136       // padded LDS row stride in bf16 elems (272B: 16B aligned, bank-conflict free)

// ---------------- node importance ----------------
__global__ void k_segmax(const float* __restrict__ ni, const long long* __restrict__ batch,
                         unsigned int* __restrict__ gmax_bits, int n)
{
    int i = blockIdx.x * blockDim.x + threadIdx.x;
    if (i >= n) return;
    // node_imp is uniform(0,1) >= 0, so float bits compare correctly as unsigned.
    atomicMax(&gmax_bits[(int)batch[i]], __float_as_uint(ni[i]));
}

__global__ void k_imp(const float* __restrict__ ni, const long long* __restrict__ batch,
                      const unsigned int* __restrict__ gmax_bits, float* __restrict__ imp, int n)
{
    int i = blockIdx.x * blockDim.x + threadIdx.x;
    if (i >= n) return;
    float g = __uint_as_float(gmax_bits[(int)batch[i]]);
    imp[i] = ni[i] / (g * 10.0f) + 0.9f;
}

// ---------------- edge scatter-add (GIN aggregation) ----------------
// 32 threads per edge, float4 per lane -> 128 features. Gather from h hits 192MB L2.
__global__ void k_edge_agg(const float* __restrict__ h, const long long* __restrict__ ei,
                           float* __restrict__ agg, long long nE)
{
    long long t = (long long)blockIdx.x * blockDim.x + threadIdx.x;
    long long e = t >> 5;
    if (e >= nE) return;
    int lane = (int)(t & 31);
    long long s = ei[e];
    long long d = ei[nE + e];
    const float4 v = *(const float4*)(h + s * DIM + lane * 4);
    float* ap = agg + d * DIM + lane * 4;
    atomicAdd(ap + 0, v.x);
    atomicAdd(ap + 1, v.y);
    atomicAdd(ap + 2, v.z);
    atomicAdd(ap + 3, v.w);
}

// ---------------- GEMM1: t = relu((agg + 1.1*h) @ W1 + b1) -> bf16 ----------------
// One wave owns a 16-row x 128-col tile. 8 waves/block = 128 rows.
__launch_bounds__(256)
__global__ void k_gemm1(const float* __restrict__ hin, const float* __restrict__ agg,
                        const float* __restrict__ W1, const float* __restrict__ b1,
                        bf16_t* __restrict__ tout, int n)
{
    __shared__ bf16_t sW[DIM * WPAD];   // W1 transposed: sW[col][k]
    __shared__ float  sB[DIM];

    const int tid = threadIdx.x;
    for (int idx = tid; idx < DIM * DIM; idx += 256) {
        int k = idx >> 7, c = idx & 127;
        sW[c * WPAD + k] = (bf16_t)W1[idx];
    }
    if (tid < DIM) sB[tid] = b1[tid];
    __syncthreads();

    const int wave = tid >> 5, lane = tid & 31;
    const int half = lane >> 4, lm = lane & 15;
    const long long row0 = (long long)blockIdx.x * 128 + wave * 16;
    const long long m = row0 + lm;
    const bool mv = (m < (long long)n);
    const bool fullTile = (row0 + 16 <= (long long)n);   // wave-uniform

    // A fragments: 16-bit A 16x32 layout; per lane-half, elems 0..7 -> K=kb0.., 8..15 -> K=kb0+16..
    v16bf afrag[4];
#pragma unroll
    for (int kk = 0; kk < 4; ++kk) {
        const int kb0 = kk * 32 + half * 8;
        const int kb1 = kb0 + 16;
        float z[16];
#pragma unroll
        for (int e = 0; e < 16; ++e) z[e] = 0.f;
        if (mv) {
            const float* hp = hin + m * DIM;
            const float* ap = agg + m * DIM;
            float4 h0 = *(const float4*)(hp + kb0);
            float4 h1 = *(const float4*)(hp + kb0 + 4);
            float4 h2 = *(const float4*)(hp + kb1);
            float4 h3 = *(const float4*)(hp + kb1 + 4);
            float4 a0 = *(const float4*)(ap + kb0);
            float4 a1 = *(const float4*)(ap + kb0 + 4);
            float4 a2 = *(const float4*)(ap + kb1);
            float4 a3 = *(const float4*)(ap + kb1 + 4);
            z[0] = a0.x + GINW * h0.x;  z[1] = a0.y + GINW * h0.y;
            z[2] = a0.z + GINW * h0.z;  z[3] = a0.w + GINW * h0.w;
            z[4] = a1.x + GINW * h1.x;  z[5] = a1.y + GINW * h1.y;
            z[6] = a1.z + GINW * h1.z;  z[7] = a1.w + GINW * h1.w;
            z[8]  = a2.x + GINW * h2.x; z[9]  = a2.y + GINW * h2.y;
            z[10] = a2.z + GINW * h2.z; z[11] = a2.w + GINW * h2.w;
            z[12] = a3.x + GINW * h3.x; z[13] = a3.y + GINW * h3.y;
            z[14] = a3.z + GINW * h3.z; z[15] = a3.w + GINW * h3.w;
        }
        v16bf af;
#pragma unroll
        for (int e = 0; e < 16; ++e) af[e] = (bf16_t)z[e];
        afrag[kk] = af;
    }

#pragma unroll
    for (int ct = 0; ct < 8; ++ct) {
        const int ncol = ct * 16 + lm;
        v8f acc = {};
#pragma unroll
        for (int kk = 0; kk < 4; ++kk) {
            const int kb0 = kk * 32 + half * 8;
            v8bf blo = *(const v8bf*)(&sW[ncol * WPAD + kb0]);
            v8bf bhi = *(const v8bf*)(&sW[ncol * WPAD + kb0 + 16]);
            v16bf bfrag = __builtin_shufflevector(blo, bhi,
                0, 1, 2, 3, 4, 5, 6, 7, 8, 9, 10, 11, 12, 13, 14, 15);
            acc = __builtin_amdgcn_wmma_f32_16x16x32_bf16(
                false, afrag[kk], false, bfrag, (short)0, acc, false, false);
        }
        const float bias = sB[ncol];
        bf16_t* tp = tout + (row0 + half * 8) * DIM + ncol;
        if (fullTile) {
            // fast path: one uniform branch, no per-store exec juggling
#pragma unroll
            for (int g = 0; g < 8; ++g) {
                float v = acc[g] + bias;
                v = v > 0.f ? v : 0.f;
                tp[(long long)g * DIM] = (bf16_t)v;
            }
        } else {
#pragma unroll
            for (int g = 0; g < 8; ++g) {
                long long mr = row0 + g + half * 8;
                if (mr < (long long)n) {
                    float v = acc[g] + bias;
                    v = v > 0.f ? v : 0.f;
                    tp[(long long)g * DIM] = (bf16_t)v;
                }
            }
        }
    }
}

// ---------------- GEMM2: hpre = relu(t @ W2 + b2); accumulate BN sum/sumsq ----------------
__launch_bounds__(256)
__global__ void k_gemm2(const bf16_t* __restrict__ tin,
                        const float* __restrict__ W2, const float* __restrict__ b2,
                        float* __restrict__ hpre, float* __restrict__ bnsum,
                        float* __restrict__ bnsq, int n)
{
    __shared__ bf16_t sW[DIM * WPAD];
    __shared__ float  sB[DIM];

    const int tid = threadIdx.x;
    for (int idx = tid; idx < DIM * DIM; idx += 256) {
        int k = idx >> 7, c = idx & 127;
        sW[c * WPAD + k] = (bf16_t)W2[idx];
    }
    if (tid < DIM) sB[tid] = b2[tid];
    __syncthreads();

    const int wave = tid >> 5, lane = tid & 31;
    const int half = lane >> 4, lm = lane & 15;
    const long long row0 = (long long)blockIdx.x * 128 + wave * 16;
    const long long m = row0 + lm;
    const bool mv = (m < (long long)n);
    const bool fullTile = (row0 + 16 <= (long long)n);   // wave-uniform

    v16bf afrag[4];
#pragma unroll
    for (int kk = 0; kk < 4; ++kk) {
        const int kb0 = kk * 32 + half * 8;
        v16bf af;
        if (mv) {
            v8bf lo = *(const v8bf*)(tin + m * DIM + kb0);
            v8bf hi = *(const v8bf*)(tin + m * DIM + kb0 + 16);
            af = __builtin_shufflevector(lo, hi,
                0, 1, 2, 3, 4, 5, 6, 7, 8, 9, 10, 11, 12, 13, 14, 15);
        } else {
#pragma unroll
            for (int e = 0; e < 16; ++e) af[e] = (bf16_t)0.f;
        }
        afrag[kk] = af;
    }

#pragma unroll
    for (int ct = 0; ct < 8; ++ct) {
        const int ncol = ct * 16 + lm;
        v8f acc = {};
#pragma unroll
        for (int kk = 0; kk < 4; ++kk) {
            const int kb0 = kk * 32 + half * 8;
            v8bf blo = *(const v8bf*)(&sW[ncol * WPAD + kb0]);
            v8bf bhi = *(const v8bf*)(&sW[ncol * WPAD + kb0 + 16]);
            v16bf bfrag = __builtin_shufflevector(blo, bhi,
                0, 1, 2, 3, 4, 5, 6, 7, 8, 9, 10, 11, 12, 13, 14, 15);
            acc = __builtin_amdgcn_wmma_f32_16x16x32_bf16(
                false, afrag[kk], false, bfrag, (short)0, acc, false, false);
        }
        const float bias = sB[ncol];
        float s1 = 0.f, s2 = 0.f;
        float* hp = hpre + (row0 + half * 8) * DIM + ncol;
        if (fullTile) {
#pragma unroll
            for (int g = 0; g < 8; ++g) {
                float v = acc[g] + bias;
                v = v > 0.f ? v : 0.f;
                hp[(long long)g * DIM] = v;
                s1 += v;
                s2 += v * v;
            }
        } else {
#pragma unroll
            for (int g = 0; g < 8; ++g) {
                long long mr = row0 + g + half * 8;
                float v = acc[g] + bias;
                v = v > 0.f ? v : 0.f;
                if (mr < (long long)n) hp[(long long)g * DIM] = v;
                else v = 0.f;           // out-of-range rows must not pollute BN stats
                s1 += v;
                s2 += v * v;
            }
        }
        // combine rows 0-7 (half 0) with rows 8-15 (half 1): same ncol in lane pair (l, l+16)
        s1 += __shfl_xor(s1, 16);
        s2 += __shfl_xor(s2, 16);
        if (half == 0) {
            atomicAdd(&bnsum[ncol], s1);
            atomicAdd(&bnsq[ncol], s2);
        }
    }
}

// ---------------- BN apply + importance scale + xcat write + pooled atomics ----------------
__global__ void k_bn_apply(const float* __restrict__ hpre,
                           const float* __restrict__ bnsum, const float* __restrict__ bnsq,
                           const float* __restrict__ gamma, const float* __restrict__ beta,
                           const float* __restrict__ imp, const long long* __restrict__ batch,
                           float* __restrict__ hnext, float* __restrict__ xcat,
                           float* __restrict__ pooled, int n, int layer, int nlay)
{
    long long t = (long long)blockIdx.x * blockDim.x + threadIdx.x;
    long long i = t >> 5;
    if (i >= (long long)n) return;
    const int f0 = (int)(t & 31) * 4;
    const float inv_n = 1.0f / (float)n;

    float4 hv = *(const float4*)(hpre + i * DIM + f0);
    float4 g4 = *(const float4*)(gamma + (long long)layer * DIM + f0);
    float4 b4 = *(const float4*)(beta + (long long)layer * DIM + f0);
    float4 s1 = *(const float4*)(bnsum + f0);
    float4 s2 = *(const float4*)(bnsq + f0);
    const float im = imp[i];
    const long long grp = batch[i];

    float4 oh;
    v4f ox;
    {
        float mu = s1.x * inv_n, var = s2.x * inv_n - mu * mu;
        float v = (hv.x - mu) * rsqrtf(var + BNEPS) * g4.x + b4.x;
        oh.x = v; ox.x = v * im;
    }
    {
        float mu = s1.y * inv_n, var = s2.y * inv_n - mu * mu;
        float v = (hv.y - mu) * rsqrtf(var + BNEPS) * g4.y + b4.y;
        oh.y = v; ox.y = v * im;
    }
    {
        float mu = s1.z * inv_n, var = s2.z * inv_n - mu * mu;
        float v = (hv.z - mu) * rsqrtf(var + BNEPS) * g4.z + b4.z;
        oh.z = v; ox.z = v * im;
    }
    {
        float mu = s1.w * inv_n, var = s2.w * inv_n - mu * mu;
        float v = (hv.w - mu) * rsqrtf(var + BNEPS) * g4.w + b4.w;
        oh.w = v; ox.w = v * im;
    }

    *(float4*)(hnext + i * DIM + f0) = oh;                       // next-layer input: L2-resident
    const long long stride = (long long)DIM * nlay;
    // xcat is 128MB write-once: stream past the caches (native clang vector for the builtin)
    __builtin_nontemporal_store(ox, (v4f*)(xcat + i * stride + (long long)layer * DIM + f0));
    float* pp = pooled + grp * stride + (long long)layer * DIM + f0;
    atomicAdd(pp + 0, ox.x);
    atomicAdd(pp + 1, ox.y);
    atomicAdd(pp + 2, ox.z);
    atomicAdd(pp + 3, ox.w);
}

// ---------------- orchestration ----------------
extern "C" void kernel_launch(void* const* d_in, const int* in_sizes, int n_in,
                              void* d_out, int out_size, void* d_ws, size_t ws_size,
                              hipStream_t stream)
{
    (void)n_in; (void)out_size; (void)ws_size;
    const float*     x     = (const float*)d_in[0];
    const float*     nimp  = (const float*)d_in[1];
    const long long* ei    = (const long long*)d_in[2];
    const long long* batch = (const long long*)d_in[3];
    const float*     W1    = (const float*)d_in[4];
    const float*     b1    = (const float*)d_in[5];
    const float*     W2    = (const float*)d_in[6];
    const float*     b2    = (const float*)d_in[7];
    const float*     gamma = (const float*)d_in[8];
    const float*     beta  = (const float*)d_in[9];

    const int       n    = in_sizes[0] / DIM;
    const long long nE   = (long long)in_sizes[2] / 2;
    const int       nlay = in_sizes[5] / DIM;   // b1 is (L, D)

    float* pooled = (float*)d_out;                                  // (G, L*D)
    float* xcat   = (float*)d_out + (size_t)NGRP * DIM * nlay;      // (N, L*D)

    // workspace carve
    char* w = (char*)d_ws;
    float*        hbuf = (float*)w;        w += (size_t)n * DIM * 4;
    float*        agg  = (float*)w;        w += (size_t)n * DIM * 4;
    float*        hpre = (float*)w;        w += (size_t)n * DIM * 4;
    bf16_t*       tbuf = (bf16_t*)w;       w += (size_t)n * DIM * 2;
    float*        imp  = (float*)w;        w += (size_t)n * 4;
    unsigned int* gmax = (unsigned int*)w; w += (size_t)NGRP * 4;
    float*        bnsum = (float*)w;       w += (size_t)DIM * 4;
    float*        bnsq  = (float*)w;       w += (size_t)DIM * 4;

    (void)hipMemsetAsync(gmax, 0, NGRP * sizeof(unsigned int), stream);
    (void)hipMemsetAsync(pooled, 0, (size_t)NGRP * DIM * nlay * 4, stream);

    const int thr = 256;
    k_segmax<<<(n + thr - 1) / thr, thr, 0, stream>>>(nimp, batch, gmax, n);
    k_imp<<<(n + thr - 1) / thr, thr, 0, stream>>>(nimp, batch, gmax, imp, n);

    const float* hcur = x;
    const int gblocks = (n + 127) / 128;
    for (int l = 0; l < nlay; ++l) {
        (void)hipMemsetAsync(agg, 0, (size_t)n * DIM * 4, stream);
        (void)hipMemsetAsync(bnsum, 0, DIM * 4, stream);
        (void)hipMemsetAsync(bnsq, 0, DIM * 4, stream);

        const long long eThreads = nE * 32;
        k_edge_agg<<<(unsigned)((eThreads + thr - 1) / thr), thr, 0, stream>>>(hcur, ei, agg, nE);

        k_gemm1<<<gblocks, 256, 0, stream>>>(hcur, agg,
                                             W1 + (size_t)l * DIM * DIM, b1 + (size_t)l * DIM,
                                             tbuf, n);
        k_gemm2<<<gblocks, 256, 0, stream>>>(tbuf,
                                             W2 + (size_t)l * DIM * DIM, b2 + (size_t)l * DIM,
                                             hpre, bnsum, bnsq, n);

        const long long bThreads = (long long)n * 32;
        k_bn_apply<<<(unsigned)((bThreads + thr - 1) / thr), thr, 0, stream>>>(
            hpre, bnsum, bnsq, gamma, beta, imp, batch, hbuf, xcat, pooled, n, l, nlay);

        hcur = hbuf;
    }
}
